// GraphAttentionLayer_40759239639872
// MI455X (gfx1250) — matrix operand order repaired and checked by reading, not compiled, not run on hardware
//
#include <hip/hip_runtime.h>

// MI455X (gfx1250) GAT layer, flash-attention style with a per-row *constant*
// softmax bound (C_i = leaky_relu(wh1_i + max_j wh2_j) >= e_ij by monotonicity),
// so the hot loop has no online-max rescaling and no loop-carried serial
// dependency (4-way split partial sums for Z).
// Roofline: adj = 256 MB read exactly once (~11us @ 23.3 TB/s); att@wh =
// 8.6 GFLOPs on v_wmma_f32_16x16x32_f16 (~2us); hot loop is ~7 VALU/element.
//
// Workspace layout (d_ws, needs ~3.2 MB):
//   [0, 2MB)              wh    f32 [8192][64]
//   [2MB, 2MB+32KB)       wh1l  f32 [8192]   (wh@a1, pre-scaled by log2(e))
//   [2MB+32KB, 2MB+64KB)  wh2l  f32 [8192]   (wh@a2, pre-scaled by log2(e))
//   [2MB+64KB, +1MB)      whx   f16, pre-swizzled WMMA B-tiles (32x16 per tile)

typedef __attribute__((ext_vector_type(16))) _Float16 v16h;
typedef __attribute__((ext_vector_type(8)))  float    v8f;

#define GAT_N   8192
#define GAT_IN  128
#define GAT_OUT 64
#define GAT_ALPHA 0.2f
#define GAT_LOG2E 1.4426950408889634f

union V16H { v16h v; _Float16 e[16]; };

// ---------------------------------------------------------------------------
// Kernel 1: wh = h @ w  (M=8192, K=128, N=64), f16 WMMA with f32 accumulate.
// One wave per 16x16 output tile; block = 4 waves = 4 column tiles of one row
// tile.  A-layout (16-bit A 16x32): lane m=L&15 is row M; element e ->
// K=16*hi+e (contiguous, same order as B).  B-layout (32x16): element e -> K=e+16*hi.
// ---------------------------------------------------------------------------
__global__ __launch_bounds__(128) void gat_wh_kernel(const float* __restrict__ h,
                                                     const float* __restrict__ w,
                                                     float* __restrict__ wh) {
  const int lane = threadIdx.x & 31;
  const int cb   = threadIdx.x >> 5;           // column tile 0..3
  const int rowbase = blockIdx.x * 16;
  const int m  = lane & 15;
  const int hi = lane >> 4;

  v8f acc = {};
#pragma unroll
  for (int k0 = 0; k0 < GAT_IN; k0 += 32) {
    V16H A;
    const float* hrow = h + (size_t)(rowbase + m) * GAT_IN + k0 + 16 * hi;
#pragma unroll
    for (int e = 0; e < 8; ++e) A.e[e]     = (_Float16)hrow[e];        // K = 16hi+e
#pragma unroll
    for (int e = 0; e < 8; ++e) A.e[8 + e] = (_Float16)hrow[8 + e];    // K = 16hi+8+e

    V16H B;
#pragma unroll
    for (int e = 0; e < 16; ++e)
      B.e[e] = (_Float16)w[(size_t)(k0 + 16 * hi + e) * GAT_OUT + cb * 16 + m];

    acc = __builtin_amdgcn_wmma_f32_16x16x32_f16(false, A.v, false, B.v,
                                                 (short)0, acc, false, false);
  }
#pragma unroll
  for (int r = 0; r < 8; ++r)
    wh[(size_t)(rowbase + r + 8 * hi) * GAT_OUT + cb * 16 + m] = acc[r];
}

// ---------------------------------------------------------------------------
// Kernel 2: per-row logits wh1l/wh2l (pre-scaled by log2e so the hot loop uses
// raw v_exp_f32) + write wh as f16 in pre-swizzled WMMA B-tile layout.
// Tile (jt, cb) covers wh rows jt*32..+31 (K dim) x cols cb*16..+15; element
// (K,n): hi=(K>>3)&1, lane=(n&15)+16*hi, v=4*(K>>4)+((K&7)>>1), half=K&1; halfword offset
// = (jt*4+cb)*512 + lane*16 + v*2 + half  (32B contiguous per lane).
// ---------------------------------------------------------------------------
__global__ void gat_prep_kernel(const float* __restrict__ wh,
                                const float* __restrict__ a,
                                float* __restrict__ wh1l,
                                float* __restrict__ wh2l,
                                _Float16* __restrict__ whx) {
  const int i = blockIdx.x * blockDim.x + threadIdx.x;
  if (i >= GAT_N) return;
  const float* row = wh + (size_t)i * GAT_OUT;
  float rv[GAT_OUT];
  float s1 = 0.f, s2 = 0.f;
#pragma unroll
  for (int n = 0; n < GAT_OUT; ++n) {
    rv[n] = row[n];
    s1 += rv[n] * a[n];
    s2 += rv[n] * a[GAT_OUT + n];
  }
  wh1l[i] = s1 * GAT_LOG2E;
  wh2l[i] = s2 * GAT_LOG2E;

  const int jt = i >> 5, K = i & 31;
  const int hi = (K >> 3) & 1, v = ((K >> 4) << 2) + ((K & 7) >> 1), half = K & 1;
#pragma unroll
  for (int n = 0; n < GAT_OUT; ++n) {
    const int cbn = n >> 4, nn = n & 15, ln = nn + 16 * hi;
    whx[(size_t)(jt * 4 + cbn) * 512 + ln * 16 + v * 2 + half] = (_Float16)rv[n];
  }
}

// ---------------------------------------------------------------------------
// Kernel 3: masked softmax + att@wh + bias + ELU.  One wave per 16-row block
// (512 single-wave workgroups -> flexible WGP packing, EXEC always all-ones).
// Lane L handles row L&15; hi=L>>4 selects which K-halves it feeds into the
// WMMA A matrix (adj/wh2 runs at j0+8hi and j0+16+8hi).
// Per-row constant bound C2 = leaky(wh1l + max_all(wh2l)) (base-2 domain);
// p = exp2(leaky(wh1l+wh2l) - C2) in (0,1], masked entries -> exp2(-big) = 0.
// ---------------------------------------------------------------------------
__global__ __launch_bounds__(32) void gat_attn_kernel(const int*   __restrict__ adj,
                                                      const float* __restrict__ wh1l,
                                                      const float* __restrict__ wh2lg,
                                                      const _Float16* __restrict__ whx,
                                                      const float* __restrict__ bvec,
                                                      float* __restrict__ out) {
  __shared__ float s_wh2[GAT_N];                       // 32 KB of 320 KB LDS
  const int lane = threadIdx.x & 31;

  // Load wh2l (already *log2e) into LDS; fold in the global max reduction.
  float gmax = -3.0e38f;
  for (int c = 0; c < GAT_N / 32; ++c) {
    const float v = wh2lg[c * 32 + lane];
    s_wh2[c * 32 + lane] = v;
    gmax = fmaxf(gmax, v);
  }
#pragma unroll
  for (int off = 16; off > 0; off >>= 1)
    gmax = fmaxf(gmax, __shfl_xor(gmax, off, 32));
  __syncthreads();                                     // orders LDS writes/reads

  const int rowbase = blockIdx.x * 16;
  const int m  = lane & 15;
  const int hi = lane >> 4;
  const int srcbase = hi * 8;                          // epilogue bpermute base

  const float wh1r = wh1l[rowbase + m];
  const float cs   = wh1r + gmax;
  const float C2   = fmaxf(cs, GAT_ALPHA * cs);        // leaky commutes w/ +scale

  float bv[4];
#pragma unroll
  for (int cbn = 0; cbn < 4; ++cbn) bv[cbn] = bvec[cbn * 16 + m];

  const int* arow = adj + (size_t)(rowbase + m) * GAT_N;

  // 4-way split partial sums: no serial dependency chain across the 16 adds.
  float ls0 = 0.0f, ls1 = 0.0f, ls2 = 0.0f, ls3 = 0.0f;
  v8f acc0 = {}, acc1 = {}, acc2 = {}, acc3 = {};

  for (int jt = 0; jt < GAT_N / 32; ++jt) {
    const int j0 = jt * 32;
    // adj: two contiguous 8-dword runs per lane (32B-aligned b128 pairs)
    const int4* ap  = (const int4*)(arow + j0 + 8 * hi);
    const int4* ap2 = (const int4*)(arow + j0 + 16 + 8 * hi);
    int4 q0 = ap[0], q1 = ap[1], q2 = ap2[0], q3 = ap2[1];
    if (jt + 1 < GAT_N / 32)
      __builtin_prefetch(arow + j0 + 32, 0, 1);        // global_prefetch_b8

    int am[16] = {q0.x, q0.y, q0.z, q0.w, q1.x, q1.y, q1.z, q1.w,
                  q2.x, q2.y, q2.z, q2.w, q3.x, q3.y, q3.z, q3.w};

    const float4* wp  = (const float4*)(s_wh2 + j0 + 8 * hi);
    const float4* wp2 = (const float4*)(s_wh2 + j0 + 16 + 8 * hi);
    float4 f0 = wp[0], f1 = wp[1], f2 = wp2[0], f3 = wp2[1];
    float w2v[16] = {f0.x, f0.y, f0.z, f0.w, f1.x, f1.y, f1.z, f1.w,
                     f2.x, f2.y, f2.z, f2.w, f3.x, f3.y, f3.z, f3.w};

    // p = exp2(leaky(wh1+wh2) - C2), masked -> 0.  Straight-line, no max pass;
    // independent math per element, partial sums split 4 ways.
    float pv[16];
#pragma unroll
    for (int e = 0; e < 16; ++e) {
      const float s  = wh1r + w2v[e];
      const float l2 = fmaxf(s, GAT_ALPHA * s);        // leaky_relu (alpha<1)
      const float t  = (am[e] > 0) ? (l2 - C2) : -3.0e38f;
      pv[e] = __builtin_amdgcn_exp2f(t);               // v_exp_f32
    }
    V16H P;
#pragma unroll
    for (int e = 0; e < 16; e += 4) {
      ls0 += pv[e + 0];
      ls1 += pv[e + 1];
      ls2 += pv[e + 2];
      ls3 += pv[e + 3];
      P.e[e + 0] = (_Float16)pv[e + 0];
      P.e[e + 1] = (_Float16)pv[e + 1];
      P.e[e + 2] = (_Float16)pv[e + 2];
      P.e[e + 3] = (_Float16)pv[e + 3];
    }

    // B tiles: one contiguous 32B load per lane from pre-swizzled whx (L2-hot)
    const size_t tb = (size_t)(jt * 4) * 512;
    v16h B0 = ((const v16h*)(whx + tb + 0 * 512))[lane];
    v16h B1 = ((const v16h*)(whx + tb + 1 * 512))[lane];
    v16h B2 = ((const v16h*)(whx + tb + 2 * 512))[lane];
    v16h B3 = ((const v16h*)(whx + tb + 3 * 512))[lane];

    acc0 = __builtin_amdgcn_wmma_f32_16x16x32_f16(false, P.v, false, B0, (short)0, acc0, false, false);
    acc1 = __builtin_amdgcn_wmma_f32_16x16x32_f16(false, P.v, false, B1, (short)0, acc1, false, false);
    acc2 = __builtin_amdgcn_wmma_f32_16x16x32_f16(false, P.v, false, B2, (short)0, acc2, false, false);
    acc3 = __builtin_amdgcn_wmma_f32_16x16x32_f16(false, P.v, false, B3, (short)0, acc3, false, false);
  }

  // Epilogue: combine split sums + partner lane, normalize, bias, ELU, store.
  const float lsum = (ls0 + ls1) + (ls2 + ls3);
  const float ltot = lsum + __shfl_xor(lsum, 16, 32);
#pragma unroll
  for (int r = 0; r < 8; ++r) {
    // acc VGPR r holds row r (lanes<16) / row r+8 (lanes>=16)
    const float Zr = __shfl(ltot, srcbase + r, 32);    // ds_bpermute
    const float zinv = (Zr > 0.0f) ? (1.0f / Zr) : 0.0f;
    const int rowg = rowbase + r + 8 * hi;
    float vals[4] = {acc0[r], acc1[r], acc2[r], acc3[r]};
#pragma unroll
    for (int cbn = 0; cbn < 4; ++cbn) {
      float x = vals[cbn] * zinv + bv[cbn];
      x = (x > 0.0f) ? x : (__expf(x) - 1.0f);         // ELU
      out[(size_t)rowg * GAT_OUT + cbn * 16 + m] = x;
    }
  }
}

// ---------------------------------------------------------------------------
extern "C" void kernel_launch(void* const* d_in, const int* in_sizes, int n_in,
                              void* d_out, int out_size, void* d_ws, size_t ws_size,
                              hipStream_t stream) {
  const float* h   = (const float*)d_in[0];   // [8192,128]
  const int*   adj = (const int*)d_in[1];     // [8192,8192]
  const float* w   = (const float*)d_in[2];   // [128,64]
  const float* a   = (const float*)d_in[3];   // [128,1]
  const float* b   = (const float*)d_in[4];   // [64]
  float* out = (float*)d_out;                 // [8192,64]

  char* ws = (char*)d_ws;
  float*    wh   = (float*)ws;                                  // 2 MB
  float*    wh1l = (float*)(ws + (2u << 20));                   // 32 KB
  float*    wh2l = wh1l + GAT_N;                                // 32 KB
  _Float16* whx  = (_Float16*)(ws + (2u << 20) + (64u << 10));  // 1 MB

  // K1: wh = h@w   (512 row-tiles x 4 col-tiles; 4 waves/block)
  gat_wh_kernel<<<GAT_N / 16, 128, 0, stream>>>(h, w, wh);
  // K2: scaled logits + B-tile swizzle
  gat_prep_kernel<<<GAT_N / 256, 256, 0, stream>>>(wh, a, wh1l, wh2l, whx);
  // K3: masked softmax-attention aggregate (512 single-wave blocks x 16 rows)
  gat_attn_kernel<<<GAT_N / 16, 32, 0, stream>>>(adj, wh1l, wh2l, whx, b, out);

  (void)in_sizes; (void)n_in; (void)out_size; (void)ws_size;
}
